// DynamicGatedMultiheadAttention_89730456748227
// MI455X (gfx1250) — matrix operand order repaired and checked by reading, not compiled, not run on hardware
//
#include <hip/hip_runtime.h>
#include <hip/hip_bf16.h>

// DynamicGatedMultiheadAttention for MI455X (gfx1250).
// Key identity: TOPK == E  =>  top-k mask is all-ones  =>  every DGL layer is a
// plain linear layer. Gate params (ln_g, ln_b, gw1, gb1, gw2, gb2) are dead.
// Pipeline: bf16 pack -> QKV proj (TDM-staged weights + WMMA bf16)
//        -> 2-pass flash attention (WMMA + online softmax + attn-mean atomics)
//        -> out proj (TDM-staged weights + WMMA bf16, f32 out).

#define E_DIM 1024
#define HEADS 16
#define HDIM  64
#define TSEQ  2048
#define BSZ   2
#define NTOK  (TSEQ * BSZ)   // 4096 tokens; (T,B,E) flat == (NTOK, E) with n = t*B + b

typedef __attribute__((ext_vector_type(16))) __bf16 v16bf;
typedef __attribute__((ext_vector_type(8)))  float  v8f;
typedef __attribute__((ext_vector_type(4)))  unsigned int v4u;
typedef __attribute__((ext_vector_type(8)))  int v8i;
typedef __attribute__((ext_vector_type(4)))  int v4i;

union BF16Frag {
  v16bf v;
  uint4 q[2];
  unsigned short u[16];
};

#define WMMA_BF16(A, B, C) \
  __builtin_amdgcn_wmma_f32_16x16x32_bf16(false, (A), false, (B), (short)0, (C), false, false)

#if __has_builtin(__builtin_amdgcn_tensor_load_to_lds)
#define HAVE_TDM 1
#else
#define HAVE_TDM 0
#endif

__device__ __forceinline__ unsigned short f2bf(float x) {
  unsigned int b = __float_as_uint(x);
  b += 0x7FFFu + ((b >> 16) & 1u);    // round-to-nearest-even
  return (unsigned short)(b >> 16);
}

// A-matrix fragment (16x32 bf16). Lane holds row m = lane&15, kbase = (lane<16)?0:16.
// Element i maps to k = kbase + i: 32 contiguous bytes.
template <typename P>
__device__ __forceinline__ v16bf load_a_frag(const P* rowk /* &row[kbase] */) {
  BF16Frag f;
  f.q[0] = *(const uint4*)(rowk);
  f.q[1] = *(const uint4*)(rowk + 8);
  return f.v;
}

// B-matrix fragment (32x16 bf16). Lane holds col n = lane&15, koff = (lane<16)?0:16.
// Element i maps to k = koff + i: 32 contiguous bytes.
template <typename P>
__device__ __forceinline__ v16bf load_b_frag(const P* colk /* &col[koff] */) {
  BF16Frag f;
  f.q[0] = *(const uint4*)(colk);
  f.q[1] = *(const uint4*)(colk + 8);
  return f.v;
}

#if HAVE_TDM
// TDM: DMA a 64(row) x 128(col) bf16 tile of a row-major matrix into LDS.
// D# packing per CDNA5 ISA ch.8: group0 = {count=1 | lds_addr | global_addr | type=2},
// group1 = {data_size=1(2B), tensor_dim0=E, tensor_dim1, tile_dim0=128, tile_dim1=64,
//           tensor_dim0_stride=row_stride}. Remaining groups zero (2-D tensor).
// This toolchain's builtin takes 6 args: (v4u, v8i, v4i, v4i, v8i, i32 cpol).
__device__ __forceinline__ void tdm_load_64x128(unsigned lds_off,
                                                const unsigned short* gptr,
                                                unsigned row_stride_elems) {
  unsigned long long ga = (unsigned long long)(uintptr_t)gptr;
  v4u g0;
  g0.x = 1u;                                              // count=1, user mode
  g0.y = lds_off;                                         // LDS byte address
  g0.z = (unsigned)(ga & 0xFFFFFFFFu);                    // global_addr[31:0]
  g0.w = (unsigned)((ga >> 32) & 0x1FFFFFFu) | (2u << 30); // ga[56:32] | type=2
  v8i g1;
  g1[0] = 0x10000;                                        // data_size=1 (2 bytes)
  g1[1] = (int)(((unsigned)E_DIM & 0xFFFFu) << 16);       // tensor_dim0 lo16
  g1[2] = (int)((4096u & 0xFFFFu) << 16);                 // dim0 hi16=0 | tensor_dim1 lo16
  g1[3] = (int)(128u << 16);                              // dim1 hi16=0 | tile_dim0=128
  g1[4] = 64;                                             // tile_dim1=64 | tile_dim2=0
  g1[5] = (int)row_stride_elems;                          // tensor_dim0_stride[31:0]
  g1[6] = 0;
  g1[7] = 0;
  v4i z4 = {0, 0, 0, 0};
  v8i z8 = {0, 0, 0, 0, 0, 0, 0, 0};
  __builtin_amdgcn_tensor_load_to_lds(g0, g1, z4, z4, z8, 0);
}
#endif

// ---------------------------------------------------------------------------
// fp32 -> bf16 pack (4 elements/thread, b128 in / b64 out)
// ---------------------------------------------------------------------------
__global__ void cast_f32_to_bf16(const float* __restrict__ src,
                                 unsigned short* __restrict__ dst, int n) {
  int i = (blockIdx.x * blockDim.x + threadIdx.x) * 4;
  if (i < n) {
    float4 f = *(const float4*)(src + i);
    uint2 o;
    o.x = (unsigned)f2bf(f.x) | ((unsigned)f2bf(f.y) << 16);
    o.y = (unsigned)f2bf(f.z) | ((unsigned)f2bf(f.w) << 16);
    *(uint2*)(dst + i) = o;
  }
}

__global__ void zero_f32(float* __restrict__ p, size_t n) {
  size_t i = (size_t)blockIdx.x * blockDim.x + threadIdx.x;
  size_t stride = (size_t)gridDim.x * blockDim.x;
  for (; i < n; i += stride) p[i] = 0.0f;
}

// ---------------------------------------------------------------------------
// Shared GEMM body: one wave computes a 16(token) x 64(out) tile of x @ W.T.
// Weight K-panels (64 rows x 128 cols) are DMA'd into LDS by the TDM and
// double-buffered: issue chunk i+1, s_wait_tensorcnt 1, consume chunk i.
// A-fragments for the full 128-wide chunk are preloaded (4 reg frags) so the
// global loads batch into one clause instead of interleaving with each WMMA.
// ---------------------------------------------------------------------------
template <typename WB>
__device__ __forceinline__ void gemm_tile_16x64(
    const unsigned short* __restrict__ xrow,   // &x[(ntile+nloc)*E]
    const unsigned short* __restrict__ wrow,   // &W[obase*E]
    WB& wbuf,                                  // __shared__ ushort [2][64*128]
    int lane, v8f (&acc)[4]) {
  const int nloc  = lane & 15;
  const int hi    = lane >> 4;
  const int kbase = hi ? 16 : 0;
  const int koff  = hi ? 16 : 0;

#if HAVE_TDM
  // NOTE: generic->LDS offset truncation; TDM wants the raw LDS byte address.
  const unsigned lds0 = (unsigned)(uintptr_t)&wbuf[0][0];
  const unsigned lds1 = (unsigned)(uintptr_t)&wbuf[1][0];
  tdm_load_64x128(lds0, wrow, E_DIM);
  for (int c0 = 0; c0 < E_DIM; c0 += 128) {
    const int cur = (c0 >> 7) & 1;
    if (c0 + 128 < E_DIM) {
      tdm_load_64x128(cur ? lds0 : lds1, wrow + c0 + 128, E_DIM);
      __builtin_amdgcn_s_wait_tensorcnt(1);   // chunk `cur` landed, next in flight
    } else {
      __builtin_amdgcn_s_wait_tensorcnt(0);
    }
    asm volatile("" ::: "memory");            // fence LDS reads behind the wait
    const unsigned short* wl = &wbuf[cur][0];
    v16bf a0 = load_a_frag(xrow + c0 + 0  + kbase);
    v16bf a1 = load_a_frag(xrow + c0 + 32 + kbase);
    v16bf a2 = load_a_frag(xrow + c0 + 64 + kbase);
    v16bf a3 = load_a_frag(xrow + c0 + 96 + kbase);
    __builtin_prefetch(xrow + c0 + 128, 0, 0);  // global_prefetch_b8
#pragma unroll
    for (int j = 0; j < 4; ++j) {
      const unsigned short* bb = wl + (j * 16 + nloc) * 128 + koff;
      acc[j] = WMMA_BF16(a0, load_b_frag(bb + 0),  acc[j]);
      acc[j] = WMMA_BF16(a1, load_b_frag(bb + 32), acc[j]);
      acc[j] = WMMA_BF16(a2, load_b_frag(bb + 64), acc[j]);
      acc[j] = WMMA_BF16(a3, load_b_frag(bb + 96), acc[j]);
    }
  }
#else
  (void)wbuf;
  for (int c0 = 0; c0 < E_DIM; c0 += 32) {
    __builtin_prefetch(xrow + c0 + 512, 0, 0);
    v16bf a = load_a_frag(xrow + c0 + kbase);
#pragma unroll
    for (int j = 0; j < 4; ++j) {
      v16bf b = load_b_frag(wrow + (size_t)(j * 16 + nloc) * E_DIM + c0 + koff);
      acc[j] = WMMA_BF16(a, b, acc[j]);
    }
  }
#endif
}

// ---------------------------------------------------------------------------
// QKV projection: out = x @ W.T + b for W = in_proj_w (3E, E).
// Q/K stored head-major (bh, t, hd); V stored transposed (bh, hd, t) so the
// attention P@V B-fragments are contiguous.
// ---------------------------------------------------------------------------
__global__ __launch_bounds__(32) void proj_qkv(
    const unsigned short* __restrict__ xq, const unsigned short* __restrict__ xk,
    const unsigned short* __restrict__ xv, const unsigned short* __restrict__ w,
    const float* __restrict__ bias,
    unsigned short* __restrict__ qh, unsigned short* __restrict__ kh,
    unsigned short* __restrict__ vth) {
  __shared__ __align__(256) unsigned short wbuf[2][64 * 128];
  const int lane  = threadIdx.x & 31;
  const int ntile = blockIdx.x * 16;     // token tile
  const int obase = blockIdx.y * 64;     // output tile inside 3E
  const int which = obase >> 10;         // 0=q 1=k 2=v
  const unsigned short* x = (which == 0) ? xq : (which == 1) ? xk : xv;

  const int nloc = lane & 15;
  const int hi   = lane >> 4;

  v8f acc[4] = {};
  gemm_tile_16x64(x + (size_t)(ntile + nloc) * E_DIM,
                  w + (size_t)obase * E_DIM, wbuf, lane, acc);

#pragma unroll
  for (int j = 0; j < 4; ++j) {
    const int o    = obase + j * 16 + nloc;      // row of in_proj_w
    const float bv = bias[o];
    const int oe   = o - which * E_DIM;          // 0..E-1
    const int h    = oe >> 6;
    const int hd   = oe & 63;
#pragma unroll
    for (int r = 0; r < 8; ++r) {
      const int n  = ntile + r + 8 * hi;         // token index = t*B + b
      const int t  = n >> 1;
      const int bb = n & 1;
      const int bh = bb * HEADS + h;
      const unsigned short val = f2bf(acc[j][r] + bv);
      if (which == 0)      qh[((size_t)bh * TSEQ + t) * HDIM + hd]  = val;
      else if (which == 1) kh[((size_t)bh * TSEQ + t) * HDIM + hd]  = val;
      else                 vth[((size_t)bh * HDIM + hd) * TSEQ + t] = val;
    }
  }
}

// ---------------------------------------------------------------------------
// Attention: one wave per (16-row t-tile, head). Two-pass flash softmax:
//   pass 1: online row max / sum over s (2 WMMA per 16-col chunk)
//   pass 2: recompute scores, p = exp(s - m) / l, atomically accumulate p/H
//           into attn_weights mean, route p through LDS into A-fragments and
//           accumulate AO = P @ V with WMMA.
// ---------------------------------------------------------------------------
__global__ __launch_bounds__(32) void attn_kernel(
    const unsigned short* __restrict__ qh, const unsigned short* __restrict__ kh,
    const unsigned short* __restrict__ vth, unsigned short* __restrict__ aob,
    float* __restrict__ aw /* (B,T,T), pre-zeroed */) {
  __shared__ __align__(16) unsigned short plds[16 * 32];

  const int lane  = threadIdx.x & 31;
  const int ttile = blockIdx.x * 16;
  const int bh    = blockIdx.y;
  const int b     = bh >> 4;
  const int h     = bh & 15;
  const int nloc  = lane & 15;
  const int hi    = lane >> 4;
  const int kbase = hi ? 16 : 0;
  const int koff  = hi ? 16 : 0;
  const float scale = 0.125f;   // 1/sqrt(64)

  // Q fragments for the two 32-wide K chunks over HD=64
  const unsigned short* qrow = qh + ((size_t)bh * TSEQ + ttile + nloc) * HDIM;
  const v16bf aq0 = load_a_frag(qrow + 0 + kbase);
  const v16bf aq1 = load_a_frag(qrow + 32 + kbase);

  float rmax[8], rsum[8];
#pragma unroll
  for (int r = 0; r < 8; ++r) { rmax[r] = -1e30f; rsum[r] = 0.0f; }

  // ---- pass 1: softmax stats ----
  for (int s0 = 0; s0 < TSEQ; s0 += 16) {
    const unsigned short* krow = kh + ((size_t)bh * TSEQ + s0 + nloc) * HDIM;
    v16bf bk0 = load_b_frag(krow + 0 + koff);
    v16bf bk1 = load_b_frag(krow + 32 + koff);
    v8f c = {};
    c = WMMA_BF16(aq0, bk0, c);
    c = WMMA_BF16(aq1, bk1, c);
#pragma unroll
    for (int r = 0; r < 8; ++r) {
      float s  = c[r] * scale;
      float cm = s;
      cm = fmaxf(cm, __shfl_xor(cm, 1, 32));
      cm = fmaxf(cm, __shfl_xor(cm, 2, 32));
      cm = fmaxf(cm, __shfl_xor(cm, 4, 32));
      cm = fmaxf(cm, __shfl_xor(cm, 8, 32));   // reduce across the 16-lane half
      float nm = fmaxf(rmax[r], cm);
      float p  = __expf(s - nm);
      float ps = p;
      ps += __shfl_xor(ps, 1, 32);
      ps += __shfl_xor(ps, 2, 32);
      ps += __shfl_xor(ps, 4, 32);
      ps += __shfl_xor(ps, 8, 32);
      rsum[r] = rsum[r] * __expf(rmax[r] - nm) + ps;
      rmax[r] = nm;
    }
  }
  float rinv[8];
#pragma unroll
  for (int r = 0; r < 8; ++r) rinv[r] = 1.0f / rsum[r];

  // ---- pass 2: emit p, accumulate head-mean, AO = P @ V ----
  v8f acc[4] = {};
  for (int s0 = 0; s0 < TSEQ; s0 += 32) {
#pragma unroll
    for (int sub = 0; sub < 2; ++sub) {
      const int s = s0 + sub * 16;
      const unsigned short* krow = kh + ((size_t)bh * TSEQ + s + nloc) * HDIM;
      v16bf bk0 = load_b_frag(krow + 0 + koff);
      v16bf bk1 = load_b_frag(krow + 32 + koff);
      v8f c = {};
      c = WMMA_BF16(aq0, bk0, c);
      c = WMMA_BF16(aq1, bk1, c);
#pragma unroll
      for (int r = 0; r < 8; ++r) {
        const float p  = __expf(c[r] * scale - rmax[r]) * rinv[r];
        const int row  = r + 8 * hi;
        atomicAdd(&aw[((size_t)b * TSEQ + ttile + row) * TSEQ + s + nloc],
                  p * (1.0f / HEADS));
        plds[row * 32 + sub * 16 + nloc] = f2bf(p);
      }
    }
    __syncthreads();
    v16bf ap = load_a_frag(plds + nloc * 32 + kbase);   // ds_load_b128 x2
#pragma unroll
    for (int j = 0; j < 4; ++j) {
      const unsigned short* vcol =
          vth + ((size_t)bh * HDIM + j * 16 + nloc) * TSEQ + s0 + koff;
      v16bf bv = load_b_frag(vcol);
      acc[j] = WMMA_BF16(ap, bv, acc[j]);
    }
    __syncthreads();
  }

  // AO -> (n = t*B + b, e = h*64 + hd) bf16 for the output projection
#pragma unroll
  for (int j = 0; j < 4; ++j) {
#pragma unroll
    for (int r = 0; r < 8; ++r) {
      const int t = ttile + r + 8 * hi;
      const int e = h * HDIM + j * 16 + nloc;
      aob[((size_t)t * BSZ + b) * E_DIM + e] = f2bf(acc[j][r]);
    }
  }
}

// ---------------------------------------------------------------------------
// Output projection: attn_output = ao @ out_w.T + out_b (fp32 out to d_out).
// ---------------------------------------------------------------------------
__global__ __launch_bounds__(32) void out_proj(
    const unsigned short* __restrict__ aob, const unsigned short* __restrict__ ow,
    const float* __restrict__ bias, float* __restrict__ out) {
  __shared__ __align__(256) unsigned short wbuf[2][64 * 128];
  const int lane  = threadIdx.x & 31;
  const int ntile = blockIdx.x * 16;
  const int obase = blockIdx.y * 64;
  const int nloc  = lane & 15;
  const int hi    = lane >> 4;

  v8f acc[4] = {};
  gemm_tile_16x64(aob + (size_t)(ntile + nloc) * E_DIM,
                  ow + (size_t)obase * E_DIM, wbuf, lane, acc);

#pragma unroll
  for (int j = 0; j < 4; ++j) {
    const int o    = obase + j * 16 + nloc;
    const float bv = bias[o];
#pragma unroll
    for (int r = 0; r < 8; ++r) {
      out[(size_t)(ntile + r + 8 * hi) * E_DIM + o] = acc[j][r] + bv;
    }
  }
}

// ---------------------------------------------------------------------------
extern "C" void kernel_launch(void* const* d_in, const int* in_sizes, int n_in,
                              void* d_out, int out_size, void* d_ws, size_t ws_size,
                              hipStream_t stream) {
  (void)in_sizes; (void)n_in; (void)out_size; (void)ws_size;
  const float* q_f  = (const float*)d_in[0];
  const float* k_f  = (const float*)d_in[1];
  const float* v_f  = (const float*)d_in[2];
  const float* wio  = (const float*)d_in[3];   // (3E, E)
  const float* bio  = (const float*)d_in[4];   // (3E)
  const float* wo   = (const float*)d_in[5];   // (E, E)
  const float* bo   = (const float*)d_in[6];   // (E)
  // d_in[7..12] (ln_g, ln_b, gw1, gb1, gw2, gb2): dead — TOPK == E mask is all-ones.

  const int NE = NTOK * E_DIM;                 // 4096 * 1024

  char* ws = (char*)d_ws;
  size_t off = 0;
  auto alloc = [&](size_t bytes) -> void* {
    off = (off + 255) & ~(size_t)255;
    void* p = ws + off;
    off += bytes;
    return p;
  };
  unsigned short* xq   = (unsigned short*)alloc((size_t)NE * 2);
  unsigned short* xk   = (unsigned short*)alloc((size_t)NE * 2);
  unsigned short* xv   = (unsigned short*)alloc((size_t)NE * 2);
  unsigned short* wqkv = (unsigned short*)alloc((size_t)3 * E_DIM * E_DIM * 2);
  unsigned short* owb  = (unsigned short*)alloc((size_t)E_DIM * E_DIM * 2);
  unsigned short* qh   = (unsigned short*)alloc((size_t)NE * 2);
  unsigned short* kh   = (unsigned short*)alloc((size_t)NE * 2);
  unsigned short* vth  = (unsigned short*)alloc((size_t)NE * 2);
  unsigned short* aob  = (unsigned short*)alloc((size_t)NE * 2);

  float* attn_out = (float*)d_out;             // (T,B,E) = NE floats
  float* attn_w   = (float*)d_out + NE;        // (B,T,T)

  // pack fp32 -> bf16
  cast_f32_to_bf16<<<NE / 4 / 256, 256, 0, stream>>>(q_f, xq, NE);
  cast_f32_to_bf16<<<NE / 4 / 256, 256, 0, stream>>>(k_f, xk, NE);
  cast_f32_to_bf16<<<NE / 4 / 256, 256, 0, stream>>>(v_f, xv, NE);
  cast_f32_to_bf16<<<3 * E_DIM * E_DIM / 4 / 256, 256, 0, stream>>>(wio, wqkv, 3 * E_DIM * E_DIM);
  cast_f32_to_bf16<<<E_DIM * E_DIM / 4 / 256, 256, 0, stream>>>(wo, owb, E_DIM * E_DIM);

  // zero attn_weights mean accumulator (d_out is poisoned)
  zero_f32<<<2048, 256, 0, stream>>>(attn_w, (size_t)BSZ * TSEQ * TSEQ);

  // QKV projection: grid (token tiles, 3E/64 output tiles)
  proj_qkv<<<dim3(NTOK / 16, (3 * E_DIM) / 64), 32, 0, stream>>>(
      xq, xk, xv, wqkv, bio, qh, kh, vth);

  // attention: grid (T/16 row tiles, B*H heads)
  attn_kernel<<<dim3(TSEQ / 16, BSZ * HEADS), 32, 0, stream>>>(qh, kh, vth, aob, attn_w);

  // output projection
  out_proj<<<dim3(NTOK / 16, E_DIM / 64), 32, 0, stream>>>(aob, owb, bo, attn_out);
}
